// MoEDecoder_67654324846797
// MI455X (gfx1250) — compile-verified, matrix operand order checked by reading, not run on hardware
//
#include <hip/hip_runtime.h>

typedef __attribute__((ext_vector_type(16))) _Float16 v16h;
typedef __attribute__((ext_vector_type(8)))  _Float16 v8h;
typedef __attribute__((ext_vector_type(8)))  float    v8f;

union ABfrag { v16h v; v8h h[2]; };

#define IN_CH   512
#define HID     256
#define OUT_CH  256
#define NEXP    8
#define NTOK    65536

#define WAVES        4
#define TOK_PER_WG   32          // 2 M-tiles of 16 tokens, shared by all 4 waves
// each wave owns 4 N-tiles (64 of the 256 output columns)

// workspace offsets in halves (f16). Total 2,297,856 halves = ~4.6 MB.
#define OFF_EW1 0
#define OFF_EW2 (8*256*512)
#define OFF_EW3 (OFF_EW2 + 8*256*256)
#define OFF_GW1 (OFF_EW3 + 8*256*256)
#define OFF_GW2 (OFF_GW1 + 256*512)
#define OFF_GW3 (OFF_GW2 + 256*256)

__device__ __forceinline__ v8f wmma_f16(v16h a, v16h b, v8f c) {
    // (neg_a, A, neg_b, B, c_mod, C, reuse_a, reuse_b)
    return __builtin_amdgcn_wmma_f32_16x16x32_f16(false, a, false, b, (short)0, c,
                                                  false, false);
}

// Two 16x16 output tiles (two M-tiles vs the same B column tile).
// B fragment is loaded once and reused by both WMMAs -> 1 global b128 per wmma.
template<int K>
__device__ __forceinline__ void tile_mm2(const _Float16* ap0, const _Float16* ap1,
                                         const _Float16* bp, v8f& acc0, v8f& acc1) {
#pragma unroll
    for (int ks = 0; ks < K; ks += 32) {
        ABfrag b, a0, a1;
        b.h[0]  = *(const v8h*)(bp + ks);
        b.h[1]  = *(const v8h*)(bp + ks + 16);
        a0.h[0] = *(const v8h*)(ap0 + ks);
        a0.h[1] = *(const v8h*)(ap0 + ks + 16);
        a1.h[0] = *(const v8h*)(ap1 + ks);
        a1.h[1] = *(const v8h*)(ap1 + ks + 16);
        acc0 = wmma_f16(a0.v, b.v, acc0);
        acc1 = wmma_f16(a1.v, b.v, acc1);
    }
}

// Single-tile version (for the 16-wide gating logit layer).
template<int K>
__device__ __forceinline__ v8f tile_mm(const _Float16* ap, const _Float16* bp) {
    v8f acc = {};
#pragma unroll
    for (int ks = 0; ks < K; ks += 32) {
        ABfrag a, b;
        a.h[0] = *(const v8h*)(ap + ks);
        a.h[1] = *(const v8h*)(ap + ks + 16);
        b.h[0] = *(const v8h*)(bp + ks);
        b.h[1] = *(const v8h*)(bp + ks + 16);
        acc = wmma_f16(a.v, b.v, acc);
    }
    return acc;
}

// 32-token x 64-column layer slab for one wave:
// A [32][K] f16 (LDS, shared), Bt [256][K] f16 (global, L2-resident),
// writes relu(A*B + bias) f16 into Out [32][256] (LDS), columns wave*64..+63.
template<int K, bool RELU>
__device__ __forceinline__ void layer_slab(const _Float16* A, const _Float16* Bt,
                                           const float* bias, _Float16* Out,
                                           unsigned wave, unsigned ln, unsigned hi) {
    const unsigned kb = hi * 8u;
    const _Float16* ap0 = A + ln * K + kb;          // tokens 0..15
    const _Float16* ap1 = A + (16 + ln) * K + kb;   // tokens 16..31
    for (int nt = 0; nt < 4; ++nt) {
        const unsigned tn = wave * 4u + nt;         // global N-tile index
        const _Float16* bp = Bt + (size_t)(tn * 16 + ln) * K + kb;
        v8f acc0 = {}, acc1 = {};
        tile_mm2<K>(ap0, ap1, bp, acc0, acc1);
        float bv = bias[tn * 16 + ln];
#pragma unroll
        for (int r = 0; r < 8; ++r) {
            float v0 = acc0[r] + bv;
            float v1 = acc1[r] + bv;
            if (RELU) { v0 = fmaxf(v0, 0.0f); v1 = fmaxf(v1, 0.0f); }
            // C/D layout: lane holds rows r + 8*hi, column ln (of this tile)
            Out[(r + 8 * hi) * 256 + tn * 16 + ln]        = (_Float16)v0;
            Out[(16 + r + 8 * hi) * 256 + tn * 16 + ln]   = (_Float16)v1;
        }
    }
}

__global__ __launch_bounds__(WAVES * 32, 1)
void moe_fused(const float* __restrict__ x, float* __restrict__ out,
               const _Float16* __restrict__ ws,
               const float* __restrict__ eb1, const float* __restrict__ eb2,
               const float* __restrict__ eb3, const float* __restrict__ gb1,
               const float* __restrict__ gb2, const float* __restrict__ gb3) {
    extern __shared__ __align__(16) char smem[];
    _Float16* sX  = (_Float16*)smem;                 // 32*512 f16 = 32 KB
    _Float16* sH1 = sX  + 32 * IN_CH;                // 32*256 f16 = 16 KB
    _Float16* sH2 = sH1 + 32 * HID;                  // 32*256 f16 = 16 KB
    float* sGate  = (float*)(sH2 + 32 * HID);        // 32*8  f32 = 1 KB
    float* sLog   = sGate + 32 * NEXP;               // 32*16 f32 = 2 KB

    const unsigned tid  = threadIdx.x;
    const unsigned wave = tid >> 5;
    const unsigned lane = tid & 31u;
    const unsigned ln   = lane & 15u;
    const unsigned hi   = lane >> 4;
    const unsigned kb   = hi * 8u;

    const size_t tok0 = (size_t)blockIdx.x * TOK_PER_WG;

    // ---- cooperative load + convert 32x512 f32 token block into LDS f16 ----
    const float* xg = x + tok0 * IN_CH;
    for (int i = tid * 4; i < TOK_PER_WG * IN_CH; i += WAVES * 32 * 4) {
        float4 v = *(const float4*)(xg + i);
        sX[i + 0] = (_Float16)v.x;
        sX[i + 1] = (_Float16)v.y;
        sX[i + 2] = (_Float16)v.z;
        sX[i + 3] = (_Float16)v.w;
    }
    __syncthreads();

    // ---- gating MLP (N split across waves) ----
    layer_slab<IN_CH, true>(sX,  ws + OFF_GW1, gb1, sH1, wave, ln, hi);
    __syncthreads();
    layer_slab<HID,   true>(sH1, ws + OFF_GW2, gb2, sH2, wave, ln, hi);
    __syncthreads();
    if (wave == 0) {
        // logits: 2 M-tiles x 1 N-tile (cols 8..15 of gW3t are zero-padded)
        const _Float16* bp = ws + OFF_GW3 + (size_t)ln * HID + kb;
#pragma unroll
        for (int mt = 0; mt < 2; ++mt) {
            const _Float16* ap = sH2 + (size_t)(mt * 16 + ln) * HID + kb;
            v8f acc = tile_mm<HID>(ap, bp);
            float bv = (ln < 8u) ? gb3[ln] : 0.0f;
#pragma unroll
            for (int r = 0; r < 8; ++r)
                sLog[(mt * 16 + r + 8 * hi) * 16 + ln] = acc[r] + bv;
        }
        // softmax over 8 experts; one token per lane (32 tokens = 32 lanes)
        {
            const int t = lane;
            float mx = sLog[t * 16 + 0];
            for (int e = 1; e < NEXP; ++e) mx = fmaxf(mx, sLog[t * 16 + e]);
            float p[NEXP];
            float s = 0.0f;
            for (int e = 0; e < NEXP; ++e) {
                p[e] = __expf(sLog[t * 16 + e] - mx);
                s += p[e];
            }
            float inv = 1.0f / s;
            for (int e = 0; e < NEXP; ++e) sGate[t * NEXP + e] = p[e] * inv;
        }
    }
    __syncthreads();

    // ---- experts: fused pipeline, gated accumulation in registers ----
    // facc[nt*2 + mt]: 8 tiles = 64 VGPRs per wave
    v8f facc[8];
    v8f zero = {};
#pragma unroll
    for (int i = 0; i < 8; ++i) facc[i] = zero;

    for (int e = 0; e < NEXP; ++e) {
        layer_slab<IN_CH, true>(sX,  ws + OFF_EW1 + (size_t)e * (256 * 512),
                                eb1 + e * HID, sH1, wave, ln, hi);
        __syncthreads();
        layer_slab<HID,   true>(sH1, ws + OFF_EW2 + (size_t)e * (256 * 256),
                                eb2 + e * HID, sH2, wave, ln, hi);
        __syncthreads();

        float gm0[8], gm1[8];
#pragma unroll
        for (int r = 0; r < 8; ++r) {
            gm0[r] = sGate[(r + 8 * hi) * NEXP + e];
            gm1[r] = sGate[(16 + r + 8 * hi) * NEXP + e];
        }

        const _Float16* ap0 = sH2 + (size_t)ln * HID + kb;
        const _Float16* ap1 = sH2 + (size_t)(16 + ln) * HID + kb;
        const _Float16* w3  = ws + OFF_EW3 + (size_t)e * (256 * 256);
#pragma unroll
        for (int nt = 0; nt < 4; ++nt) {
            const unsigned tn = wave * 4u + nt;
            const _Float16* bp = w3 + (size_t)(tn * 16 + ln) * HID + kb;
            v8f acc0 = {}, acc1 = {};
            tile_mm2<HID>(ap0, ap1, bp, acc0, acc1);
            float bv = eb3[e * OUT_CH + tn * 16 + ln];
#pragma unroll
            for (int r = 0; r < 8; ++r) {
                facc[nt * 2 + 0][r] += gm0[r] * (acc0[r] + bv);
                facc[nt * 2 + 1][r] += gm1[r] * (acc1[r] + bv);
            }
        }
        __syncthreads();   // protect sH1/sH2 reuse across expert iterations
    }

    // ---- store 32x(wave's 64 cols) f32 output ----
#pragma unroll
    for (int nt = 0; nt < 4; ++nt) {
        const unsigned tn = wave * 4u + nt;
#pragma unroll
        for (int mt = 0; mt < 2; ++mt) {
#pragma unroll
            for (int r = 0; r < 8; ++r) {
                out[(tok0 + mt * 16 + r + 8 * hi) * OUT_CH + tn * 16 + ln] =
                    facc[nt * 2 + mt][r];
            }
        }
    }
}

// Weight prep: f32 [b][K][N] -> f16 transposed [b][Npad][K] (zero-pad cols >= N)
__global__ void cvt_w(const float* __restrict__ W, _Float16* __restrict__ Wt,
                      int K, int N, int Npad) {
    int b = blockIdx.y;
    int idx = blockIdx.x * blockDim.x + threadIdx.x;
    if (idx >= Npad * K) return;
    int n = idx / K;
    int k = idx - n * K;
    float v = (n < N) ? W[(size_t)b * K * N + (size_t)k * N + n] : 0.0f;
    Wt[(size_t)b * Npad * K + idx] = (_Float16)v;
}

extern "C" void kernel_launch(void* const* d_in, const int* in_sizes, int n_in,
                              void* d_out, int out_size, void* d_ws, size_t ws_size,
                              hipStream_t stream) {
    const float* x   = (const float*)d_in[0];
    const float* eW1 = (const float*)d_in[1];
    const float* eb1 = (const float*)d_in[2];
    const float* eW2 = (const float*)d_in[3];
    const float* eb2 = (const float*)d_in[4];
    const float* eW3 = (const float*)d_in[5];
    const float* eb3 = (const float*)d_in[6];
    const float* gW1 = (const float*)d_in[7];
    const float* gb1 = (const float*)d_in[8];
    const float* gW2 = (const float*)d_in[9];
    const float* gb2 = (const float*)d_in[10];
    const float* gW3 = (const float*)d_in[11];
    const float* gb3 = (const float*)d_in[12];
    float* out = (float*)d_out;
    _Float16* ws = (_Float16*)d_ws;

    dim3 blk(256);
    // eW1: 8 x [512][256] -> [256][512]
    cvt_w<<<dim3(512, 8), blk, 0, stream>>>(eW1, ws + OFF_EW1, 512, 256, 256);
    // eW2, eW3: 8 x [256][256] -> [256][256]
    cvt_w<<<dim3(256, 8), blk, 0, stream>>>(eW2, ws + OFF_EW2, 256, 256, 256);
    cvt_w<<<dim3(256, 8), blk, 0, stream>>>(eW3, ws + OFF_EW3, 256, 256, 256);
    // gating weights
    cvt_w<<<dim3(512, 1), blk, 0, stream>>>(gW1, ws + OFF_GW1, 512, 256, 256);
    cvt_w<<<dim3(256, 1), blk, 0, stream>>>(gW2, ws + OFF_GW2, 256, 256, 256);
    cvt_w<<<dim3(16, 1),  blk, 0, stream>>>(gW3, ws + OFF_GW3, 256, 8, 16);

    size_t smem = (size_t)TOK_PER_WG * IN_CH * 2   // sX  f16
                + (size_t)TOK_PER_WG * HID * 2     // sH1 f16
                + (size_t)TOK_PER_WG * HID * 2     // sH2 f16
                + (size_t)TOK_PER_WG * NEXP * 4    // sGate f32
                + (size_t)TOK_PER_WG * 16 * 4;     // sLog f32
    moe_fused<<<dim3(NTOK / TOK_PER_WG), dim3(WAVES * 32), smem, stream>>>(
        x, out, ws, eb1, eb2, eb3, gb1, gb2, gb3);
}
